// MultiHeadCrossAttention_28063316313030
// MI455X (gfx1250) — compile-verified, matrix operand order checked by reading, not run on hardware
//
#include <hip/hip_runtime.h>
#include <hip/hip_bf16.h>

typedef _Float16 v4h  __attribute__((ext_vector_type(4)));
typedef _Float16 v8h  __attribute__((ext_vector_type(8)));
typedef _Float16 v16h __attribute__((ext_vector_type(16)));
typedef float    v8f  __attribute__((ext_vector_type(8)));

#define BATCH 16384
#define CIN   1024
#define NOUT  1024

// ---------------------------------------------------------------------------
// Kernel 1a: transpose + convert W_V (fp32 [e][c]) -> WvT (f16 [c][e])
// ---------------------------------------------------------------------------
__global__ __launch_bounds__(256) void transpose_cvt(const float* __restrict__ Wv,
                                                     _Float16* __restrict__ WvT) {
    __shared__ float tile[32][33];
    const int bx = blockIdx.x & 31;   // c-tile
    const int by = blockIdx.x >> 5;   // e-tile
    const int eBase = by * 32, cBase = bx * 32;
    const int tr = threadIdx.x >> 5;  // 0..7
    const int tc = threadIdx.x & 31;  // 0..31
#pragma unroll
    for (int rr = 0; rr < 32; rr += 8)
        tile[tr + rr][tc] = Wv[(size_t)(eBase + tr + rr) * CIN + cBase + tc];
    __syncthreads();
#pragma unroll
    for (int rr = 0; rr < 32; rr += 8)
        WvT[(size_t)(cBase + tr + rr) * CIN + eBase + tc] = (_Float16)tile[tc][tr + rr];
}

// ---------------------------------------------------------------------------
// Fragment helper: 16-bit A/B fragment from a row-major f16 buffer.
// CDNA5 K layout: lanes 0-15 take K{0..7, 16..23}, lanes 16-31 take K{8..15, 24..31}.
// ---------------------------------------------------------------------------
__device__ inline v16h load_frag_global(const _Float16* bp, int hi) {
    union { v16h v; v8h h[2]; } u;
    u.h[0] = *(const v8h*)(bp + hi * 8);
    u.h[1] = *(const v8h*)(bp + 16 + hi * 8);
    return u.v;
}

// ---------------------------------------------------------------------------
// Kernel 1b: Wc[o][c] = sum_e Wfc[o][e] * WvT[c][e]   (f32 accum, f16 out)
// 256 WGs: 16 o-rows x 256 c-cols each (wave = 32 cols, 2 col-tiles) so the
// whole device participates instead of 64 WGs.
// ---------------------------------------------------------------------------
__global__ __launch_bounds__(256) void weight_gemm(const float* __restrict__ Wfc,
                                                   const _Float16* __restrict__ WvT,
                                                   _Float16* __restrict__ Wc) {
    __shared__ _Float16 stage[2][16][32];
    const int t = threadIdx.x, wave = t >> 5, lane = t & 31;
    const int lrow = lane & 15, hi = lane >> 4;
    const int oBase   = (blockIdx.x >> 2) * 16;            // 64 o-blocks
    const int colBase = (blockIdx.x & 3) * 256 + wave * 32; // 4 c-blocks x 8 waves

    v8f acc[2] = {};

    auto stage_fn = [&](int kb, int buf) {
        const int row = t >> 4;            // 16 rows
        const int k2 = (t & 15) << 1;      // 0..30
        const float2 f = *(const float2*)(Wfc + (size_t)(oBase + row) * CIN + kb * 32 + k2);
        stage[buf][row][k2]     = (_Float16)f.x;
        stage[buf][row][k2 + 1] = (_Float16)f.y;
    };

    stage_fn(0, 0);
    __syncthreads();
    for (int kb = 0; kb < 32; ++kb) {
        const int cur = kb & 1;
        if (kb + 1 < 32) stage_fn(kb + 1, cur ^ 1);
        v16h b[2];
#pragma unroll
        for (int j = 0; j < 2; ++j) {
            const _Float16* bp = WvT + (size_t)(colBase + j * 16 + lrow) * CIN + kb * 32;
            b[j] = load_frag_global(bp, hi);
        }
        union { v16h v; v8h h[2]; } ua;
        ua.h[0] = *(const v8h*)&stage[cur][lrow][hi * 8];
        ua.h[1] = *(const v8h*)&stage[cur][lrow][16 + hi * 8];
        const v16h a = ua.v;
#pragma unroll
        for (int j = 0; j < 2; ++j)
            acc[j] = __builtin_amdgcn_wmma_f32_16x16x32_f16(false, a, false, b[j],
                                                            (short)0, acc[j], false, false);
        __syncthreads();
    }
#pragma unroll
    for (int j = 0; j < 2; ++j)
#pragma unroll
        for (int v = 0; v < 8; ++v)
            Wc[(size_t)(oBase + v + 8 * hi) * CIN + colBase + j * 16 + lrow] = (_Float16)acc[j][v];
}

// ---------------------------------------------------------------------------
// Kernel 2: z_i = x_i @ Wc^T (i=1,2), per-row LayerNorm of each, summed:
//   out = (norm1 + norm2) * gamma + 2*beta
// WG = 512 threads (16 waves): 32 batch rows (2 row-tiles) x 1024 cols for
// BOTH inputs; each wave owns 64 cols (4 col-tiles). Each B fragment feeds
// 4 WMMAs (2 inputs x 2 row-tiles).
// ---------------------------------------------------------------------------
__global__ __launch_bounds__(512) void fused_gemm_ln(const float* __restrict__ x1,
                                                     const float* __restrict__ x2,
                                                     const _Float16* __restrict__ Wc,
                                                     const float* __restrict__ gamma,
                                                     const float* __restrict__ beta,
                                                     float* __restrict__ out) {
    __shared__ _Float16 stage[2][2][32][32];   // [buf][input][row][k]      8 KB
    __shared__ float red[16][64][2];           // [wave][input*32+row][s,q]  8 KB
    __shared__ float stats[64][2];             // [input*32+row][mu, rstd]

    const int t = threadIdx.x, wave = t >> 5, lane = t & 31;
    const int lrow = lane & 15, hi = lane >> 4;
    const int rowBase = blockIdx.x * 32;
    const int colBase = wave * 64;

    v8f acc[2][2][4] = {};                     // [input][rowtile][coltile]
    const float* xin[2] = { x1, x2 };

    auto stage_fn = [&](int kb, int buf) {
        const int i   = t >> 8;            // input 0/1 (256 threads each)
        const int s   = t & 255;
        const int row = s >> 3;            // 32 rows
        const int k4  = (s & 7) << 2;      // 0..28 step 4
        const float4 f = *(const float4*)(xin[i] + (size_t)(rowBase + row) * CIN + kb * 32 + k4);
        v4h h = { (_Float16)f.x, (_Float16)f.y, (_Float16)f.z, (_Float16)f.w };
        *(v4h*)&stage[buf][i][row][k4] = h;
    };

    stage_fn(0, 0);
    __syncthreads();
    for (int kb = 0; kb < 32; ++kb) {
        const int cur = kb & 1;
        if (kb + 1 < 32) stage_fn(kb + 1, cur ^ 1);

        // 4 B fragments (distinct registers -> one wide load clause).
        const _Float16* bbase = Wc + (size_t)(colBase + lrow) * CIN + kb * 32;
        v16h b[4];
#pragma unroll
        for (int j = 0; j < 4; ++j)
            b[j] = load_frag_global(bbase + (size_t)j * 16 * CIN, hi);

        // Prefetch next k-step's B fragments into the WGP cache.
        if (kb + 1 < 32) {
#pragma unroll
            for (int j = 0; j < 4; ++j)
                __builtin_prefetch(bbase + (size_t)j * 16 * CIN + 32, 0, 1);
        }

        v16h a[2][2];                      // [input][rowtile]
#pragma unroll
        for (int i = 0; i < 2; ++i)
#pragma unroll
            for (int rt = 0; rt < 2; ++rt) {
                union { v16h v; v8h h[2]; } u;
                u.h[0] = *(const v8h*)&stage[cur][i][rt * 16 + lrow][hi * 8];
                u.h[1] = *(const v8h*)&stage[cur][i][rt * 16 + lrow][16 + hi * 8];
                a[i][rt] = u.v;
            }
#pragma unroll
        for (int j = 0; j < 4; ++j)
#pragma unroll
            for (int i = 0; i < 2; ++i)
#pragma unroll
                for (int rt = 0; rt < 2; ++rt)
                    acc[i][rt][j] = __builtin_amdgcn_wmma_f32_16x16x32_f16(
                        false, a[i][rt], false, b[j], (short)0, acc[i][rt][j], false, false);
        __syncthreads();
    }

    // ---- per-row partial sum / sumsq (this wave's 64 cols), half-wave reduce ----
#pragma unroll
    for (int r = 0; r < 2; ++r)
#pragma unroll
        for (int rt = 0; rt < 2; ++rt)
#pragma unroll
            for (int v = 0; v < 8; ++v) {
                float s = 0.f, q = 0.f;
#pragma unroll
                for (int j = 0; j < 4; ++j) { const float z = acc[r][rt][j][v]; s += z; q += z * z; }
#pragma unroll
                for (int m = 1; m < 16; m <<= 1) {
                    s += __shfl_xor(s, m, 32);
                    q += __shfl_xor(q, m, 32);
                }
                if (lrow == 0) {   // lane 0 -> row v, lane 16 -> row v+8
                    const int row32 = rt * 16 + v + 8 * hi;
                    red[wave][r * 32 + row32][0] = s;
                    red[wave][r * 32 + row32][1] = q;
                }
            }
    __syncthreads();
    if (t < 64) {
        float s = 0.f, q = 0.f;
#pragma unroll
        for (int w = 0; w < 16; ++w) { s += red[w][t][0]; q += red[w][t][1]; }
        const float mu  = s * (1.0f / 1024.0f);
        const float var = q * (1.0f / 1024.0f) - mu * mu;
        stats[t][0] = mu;
        stats[t][1] = rsqrtf(var + 1e-5f);
    }
    __syncthreads();

    // ---- normalize, combine both inputs, apply affine, store ----
#pragma unroll
    for (int j = 0; j < 4; ++j) {
        const int col = colBase + j * 16 + lrow;
        const float g  = gamma[col];
        const float be = beta[col];
#pragma unroll
        for (int rt = 0; rt < 2; ++rt)
#pragma unroll
            for (int v = 0; v < 8; ++v) {
                const int row32 = rt * 16 + v + 8 * hi;
                const float n1 = (acc[0][rt][j][v] - stats[row32][0])      * stats[row32][1];
                const float n2 = (acc[1][rt][j][v] - stats[32 + row32][0]) * stats[32 + row32][1];
                out[(size_t)(rowBase + row32) * NOUT + col] = (n1 + n2) * g + 2.0f * be;
            }
    }
}

// ---------------------------------------------------------------------------
extern "C" void kernel_launch(void* const* d_in, const int* in_sizes, int n_in,
                              void* d_out, int out_size, void* d_ws, size_t ws_size,
                              hipStream_t stream) {
    (void)in_sizes; (void)n_in; (void)out_size; (void)ws_size;
    const float* x1    = (const float*)d_in[0];
    const float* x2    = (const float*)d_in[1];
    // d_in[2] = W_Q, d_in[3] = W_K : dead (softmax over length-1 axis == 1)
    const float* Wv    = (const float*)d_in[4];   // (H*D, C) = (1024,1024)
    const float* Wfc   = (const float*)d_in[5];   // (OUT, H*D) = (1024,1024)
    const float* gamma = (const float*)d_in[6];
    const float* beta  = (const float*)d_in[7];
    float* out = (float*)d_out;

    _Float16* WvT = (_Float16*)d_ws;              // 2 MB
    _Float16* Wc  = WvT + (size_t)CIN * CIN;      // 2 MB

    transpose_cvt<<<1024, 256, 0, stream>>>(Wv, WvT);
    weight_gemm<<<256, 256, 0, stream>>>(Wfc, WvT, Wc);
    fused_gemm_ln<<<BATCH / 32, 512, 0, stream>>>(x1, x2, Wc, gamma, beta, out);
}
